// model_4080218931292
// MI455X (gfx1250) — matrix operand order repaired and checked by reading, not compiled, not run on hardware
//
#include <hip/hip_runtime.h>
#include <cstdint>
#include <cstddef>

#define Bsz   2048
#define OBSN  20
#define PREDN 30
#define Fdim  16
#define Hdim  1024
#define FourH 4096

typedef __attribute__((ext_vector_type(16))) __bf16 v16bf;
typedef __attribute__((ext_vector_type(8)))  __bf16 v8bf;
typedef __attribute__((ext_vector_type(8)))  float  v8f;

// ---------- scalar helpers ----------
__device__ inline __bf16 f2bf(float f) {
  union { float f; uint32_t u; } v; v.f = f;
  uint32_t r = v.u + 0x7fffu + ((v.u >> 16) & 1u);   // round-to-nearest-even
  uint16_t h = (uint16_t)(r >> 16);
  __bf16 o; __builtin_memcpy(&o, &h, 2); return o;
}
__device__ inline float bf2f(__bf16 b) {
  uint16_t h; __builtin_memcpy(&h, &b, 2);
  union { uint32_t u; float f; } v; v.u = ((uint32_t)h) << 16; return v.f;
}
__device__ inline float sigmoidf_(float x) { return 1.f / (1.f + __expf(-x)); }
__device__ inline float tanhf_(float x) {
  float ax = fabsf(x);
  float e  = __expf(-2.f * ax);
  float t  = (1.f - e) / (1.f + e);
  return x < 0.f ? -t : t;
}

// ---------- fused LSTM step: gates = X@Wih^T + Hprev@Whh^T + bias; pointwise update ----------
// Block: 256 threads (8 waves). Block tile: 128 batch rows x 32 hidden cols
//   => computes the 128x128 gate slice {i,f,g,o} x 32 cols, K accumulated by WMMA bf16.
// Wave w: wm=w&3 -> 32-row batch slice, wh=w>>2 -> 16-col hidden slice; all 4 gates local.
__global__ __launch_bounds__(256) void lstm_step(
    const __bf16* __restrict__ X,     int ldx,            // [B, ldx], ldx in {32, 1024}
    const __bf16* __restrict__ Wih,                       // [4H, ldx]
    const __bf16* __restrict__ Hprev,                     // [B, H]
    const float*  __restrict__ Cprev,                     // [B, H]
    const __bf16* __restrict__ Whh,                       // [4H, H]
    const float*  __restrict__ bias,                      // [4H] (bih+bhh)
    __bf16* __restrict__ Hout,                            // [B, H]
    float*  __restrict__ Cout)                            // [B, H]
{
  __shared__ __bf16 As[128][40];   // 40-elem stride: 80B rows, 16B aligned, bank-conflict-free
  __shared__ __bf16 Bs[128][40];

  const int tid  = threadIdx.x;
  const int lane = tid & 31;
  const int w    = tid >> 5;
  const int wm   = w & 3;
  const int wh   = w >> 2;
  const int m0   = blockIdx.x * 128;
  const int hc0  = blockIdx.y * 32;

  v8f acc[4][2];
#pragma unroll
  for (int g = 0; g < 4; ++g)
#pragma unroll
    for (int mi = 0; mi < 2; ++mi)
#pragma unroll
      for (int r = 0; r < 8; ++r) acc[g][mi][r] = 0.f;

  const int ldr   = tid >> 2;          // loader row 0..63
  const int chunk = (tid & 3) * 8;     // 8 bf16 (16B) chunks

#pragma unroll
  for (int phase = 0; phase < 2; ++phase) {
    const __bf16* A = phase ? Hprev : X;
    const __bf16* W = phase ? Whh   : Wih;
    const int lda   = phase ? Hdim  : ldx;   // lda == K == leading dim of both A and W

    for (int k0 = 0; k0 < lda; k0 += 32) {
#pragma unroll
      for (int it = 0; it < 2; ++it) {
        const int r = ldr + it * 64;
        *(uint4*)(&As[r][chunk]) =
            *(const uint4*)(A + (size_t)(m0 + r) * lda + k0 + chunk);
        const int grow = ((r >> 5) << 10) + hc0 + (r & 31);   // gate*H + hc0 + j
        *(uint4*)(&Bs[r][chunk]) =
            *(const uint4*)(W + (size_t)grow * lda + k0 + chunk);
      }
      if (k0 + 32 < lda) {   // L2-resident; prefetch next K tile
        __builtin_prefetch(A + (size_t)(m0 + ldr) * lda + k0 + 32 + chunk, 0, 1);
        __builtin_prefetch(W + (size_t)(((ldr >> 5) << 10) + hc0 + (ldr & 31)) * lda + k0 + 32 + chunk, 0, 1);
      }
      __syncthreads();

      // A fragments: lane l holds M = base+(l&15); K = kh..kh+7 and kh+16..kh+23, kh = (l>=16)?8:0
      v16bf afrag[2], bfrag[4];
      const int arow = wm * 32 + (lane & 15);
      const int kh   = (lane >> 4) * 8;
#pragma unroll
      for (int mi = 0; mi < 2; ++mi) {
        const __bf16* p = &As[arow + mi * 16][kh];
        v8bf lo = *(const v8bf*)p;
        v8bf hi = *(const v8bf*)(p + 16);
        afrag[mi] = __builtin_shufflevector(lo, hi, 0,1,2,3,4,5,6,7,8,9,10,11,12,13,14,15);
      }
      // B fragments: lane l holds column N = (l&15); K = kb..kb+7 and kb+16..kb+23, kb = (l>=16)?8:0
      const int bkoff = (lane >> 4) * 8;
#pragma unroll
      for (int g = 0; g < 4; ++g) {
        const __bf16* p = &Bs[g * 32 + wh * 16 + (lane & 15)][bkoff];
        v8bf lo = *(const v8bf*)p;
        v8bf hi = *(const v8bf*)(p + 16);
        bfrag[g] = __builtin_shufflevector(lo, hi, 0,1,2,3,4,5,6,7,8,9,10,11,12,13,14,15);
      }
#pragma unroll
      for (int g = 0; g < 4; ++g)
#pragma unroll
        for (int mi = 0; mi < 2; ++mi)
          acc[g][mi] = __builtin_amdgcn_wmma_f32_16x16x32_bf16(
              false, afrag[mi], false, bfrag[g], (short)0, acc[g][mi], false, false);
      __syncthreads();
    }
  }

  // LSTM pointwise (gates all wave-local)
  const int col   = hc0 + wh * 16 + (lane & 15);
  const float b_i = bias[col];
  const float b_f = bias[Hdim + col];
  const float b_g = bias[2 * Hdim + col];
  const float b_o = bias[3 * Hdim + col];
  const int rowadd = (lane >> 4) << 3;   // C layout: M = r + 8*(lane/16)

#pragma unroll
  for (int mi = 0; mi < 2; ++mi) {
#pragma unroll
    for (int r = 0; r < 8; ++r) {
      const int row   = m0 + wm * 32 + mi * 16 + r + rowadd;
      const size_t ix = (size_t)row * Hdim + col;
      const float gi  = sigmoidf_(acc[0][mi][r] + b_i);
      const float gf  = sigmoidf_(acc[1][mi][r] + b_f);
      const float gg  = tanhf_(acc[2][mi][r] + b_g);
      const float go  = sigmoidf_(acc[3][mi][r] + b_o);
      const float c   = gf * Cprev[ix] + gi * gg;
      Cout[ix] = c;
      Hout[ix] = f2bf(go * tanhf_(c));
    }
  }
}

// ---------- small helper kernels ----------
__global__ void k_zero_u32(uint32_t* __restrict__ p, int n) {
  int i = blockIdx.x * blockDim.x + threadIdx.x;
  if (i < n) p[i] = 0u;
}
__global__ void k_cvt(__bf16* __restrict__ d, const float* __restrict__ s, int n) {
  int i = blockIdx.x * blockDim.x + threadIdx.x;
  if (i < n) d[i] = f2bf(s[i]);
}
__global__ void k_pad_wih(__bf16* __restrict__ d, const float* __restrict__ s) {
  int i = blockIdx.x * blockDim.x + threadIdx.x;   // FourH*32
  int c = i & 31, r = i >> 5;
  d[i] = (c < Fdim) ? f2bf(s[r * Fdim + c]) : f2bf(0.f);
}
__global__ void k_bias2(float* __restrict__ d, const float* __restrict__ a, const float* __restrict__ b) {
  int i = blockIdx.x * blockDim.x + threadIdx.x;   // FourH
  d[i] = a[i] + b[i];
}
// vseq[t][b][c] = v_in[b][t][c] (pad to 32); aseq: diff of consecutive timesteps
__global__ void k_build_seq(__bf16* __restrict__ d, const float* __restrict__ vin, int diff) {
  int i = blockIdx.x * blockDim.x + threadIdx.x;
  int c = i & 31; int b = (i >> 5) & (Bsz - 1); int t = i >> 16;   // Bsz*32 = 2^16
  float val = 0.f;
  if (c < Fdim) {
    size_t base = ((size_t)b * OBSN + t) * Fdim + c;
    val = diff ? (vin[base + Fdim] - vin[base]) : vin[base];
  }
  d[i] = f2bf(val);
}
__global__ void k_transpose_y(const float* __restrict__ y, const float* __restrict__ sx, float* __restrict__ o) {
  int i = blockIdx.x * blockDim.x + threadIdx.x;   // PRED*B*16
  int f = i & 15; int b = (i >> 4) & (Bsz - 1); int p = i >> 15;   // B*16 = 2^15
  o[i] = sx[0] * y[((size_t)b * PREDN + p) * Fdim + f];
}

__device__ inline float head_dot(const __bf16* __restrict__ hp, const float* __restrict__ wp) {
  float s = 0.f;
#pragma unroll 8
  for (int k = 0; k < Hdim; ++k) s = fmaf(bf2f(hp[k]), wp[k], s);
  return s;
}
__global__ __launch_bounds__(256) void k_head_v(
    const __bf16* __restrict__ h, const float* __restrict__ Wl, const float* __restrict__ bl,
    const float* __restrict__ sv, float* __restrict__ vst)
{
  int i = blockIdx.x * 256 + threadIdx.x;   // B*16
  int f = i & 15, b = i >> 4;
  float s = head_dot(h + (size_t)b * Hdim, Wl + (size_t)f * Hdim);
  vst[i] = sv[0] * tanhf_(s + bl[f]);
}
__global__ __launch_bounds__(256) void k_head_a_init(
    const __bf16* __restrict__ h, const float* __restrict__ Wl, const float* __restrict__ bl,
    const float* __restrict__ sv, const float* __restrict__ sx, const float* __restrict__ x_in,
    float* __restrict__ vst, float* __restrict__ xst, float* __restrict__ out0,
    __bf16* __restrict__ apad)
{
  int i = blockIdx.x * 256 + threadIdx.x;
  int f = i & 15, b = i >> 4;
  float a = tanhf_(head_dot(h + (size_t)b * Hdim, Wl + (size_t)f * Hdim) + bl[f]);
  float v = vst[i] + sv[0] * a;
  float x = sx[0] * x_in[((size_t)b * OBSN + (OBSN - 1)) * Fdim + f] + v;
  vst[i] = v; xst[i] = x;
  out0[i] = x;
  apad[(b << 5) + f] = f2bf(a);
}
__global__ __launch_bounds__(256) void k_head_roll(
    const __bf16* __restrict__ h, const float* __restrict__ Wl, const float* __restrict__ bl,
    const float* __restrict__ sv,
    float* __restrict__ vst, float* __restrict__ xst, float* __restrict__ outs,
    __bf16* __restrict__ apad)
{
  int i = blockIdx.x * 256 + threadIdx.x;
  int f = i & 15, b = i >> 4;
  float a = tanhf_(head_dot(h + (size_t)b * Hdim, Wl + (size_t)f * Hdim) + bl[f]);
  float v = vst[i] + sv[0] * a;
  float x = xst[i] + v;
  vst[i] = v; xst[i] = x;
  outs[i] = x;
  apad[(b << 5) + f] = f2bf(a);
}

// ---------- host launcher ----------
extern "C" void kernel_launch(void* const* d_in, const int* in_sizes, int n_in,
                              void* d_out, int out_size, void* d_ws, size_t ws_size,
                              hipStream_t stream) {
  (void)in_sizes; (void)n_in; (void)out_size; (void)ws_size;
  const float* x_in  = (const float*)d_in[0];
  const float* y     = (const float*)d_in[1];
  const float* v_in  = (const float*)d_in[2];
  const float* sx    = (const float*)d_in[3];
  const float* sv    = (const float*)d_in[4];
  const float* vWih0 = (const float*)d_in[5];
  const float* vWhh0 = (const float*)d_in[6];
  const float* vbih0 = (const float*)d_in[7];
  const float* vbhh0 = (const float*)d_in[8];
  const float* vWih1 = (const float*)d_in[9];
  const float* vWhh1 = (const float*)d_in[10];
  const float* vbih1 = (const float*)d_in[11];
  const float* vbhh1 = (const float*)d_in[12];
  const float* vWl   = (const float*)d_in[13];
  const float* vbl   = (const float*)d_in[14];
  const float* aWih0 = (const float*)d_in[15];
  const float* aWhh0 = (const float*)d_in[16];
  const float* abih0 = (const float*)d_in[17];
  const float* abhh0 = (const float*)d_in[18];
  const float* aWih1 = (const float*)d_in[19];
  const float* aWhh1 = (const float*)d_in[20];
  const float* abih1 = (const float*)d_in[21];
  const float* abhh1 = (const float*)d_in[22];
  const float* aWl   = (const float*)d_in[23];
  const float* abl   = (const float*)d_in[24];

  char* ws = (char*)d_ws;
  size_t off = 0;
  auto carve = [&](size_t bytes) -> void* {
    void* p = ws + off;
    off = (off + bytes + 255) & ~(size_t)255;
    return p;
  };

  const size_t WB = (size_t)FourH * Hdim * sizeof(__bf16);   // 8MB
  __bf16* vWhh0b = (__bf16*)carve(WB);
  __bf16* vWih1b = (__bf16*)carve(WB);
  __bf16* vWhh1b = (__bf16*)carve(WB);
  __bf16* aWhh0b = (__bf16*)carve(WB);
  __bf16* aWih1b = (__bf16*)carve(WB);
  __bf16* aWhh1b = (__bf16*)carve(WB);
  __bf16* vWih0p = (__bf16*)carve((size_t)FourH * 32 * sizeof(__bf16));
  __bf16* aWih0p = (__bf16*)carve((size_t)FourH * 32 * sizeof(__bf16));
  float*  vb0    = (float*)carve(FourH * sizeof(float));
  float*  vb1    = (float*)carve(FourH * sizeof(float));
  float*  ab0    = (float*)carve(FourH * sizeof(float));
  float*  ab1    = (float*)carve(FourH * sizeof(float));
  __bf16* vseq   = (__bf16*)carve((size_t)OBSN * Bsz * 32 * sizeof(__bf16));
  __bf16* aseq   = (__bf16*)carve((size_t)(OBSN - 1) * Bsz * 32 * sizeof(__bf16));
  const size_t HB = (size_t)Bsz * Hdim * sizeof(__bf16);     // 4MB
  const size_t CB = (size_t)Bsz * Hdim * sizeof(float);      // 8MB
  __bf16* hz = (__bf16*)carve(HB);
  float*  cz = (float*)carve(CB);
  __bf16* h0p[2] = { (__bf16*)carve(HB), (__bf16*)carve(HB) };
  float*  c0p[2] = { (float*)carve(CB),  (float*)carve(CB)  };
  __bf16* h1p[2] = { (__bf16*)carve(HB), (__bf16*)carve(HB) };
  float*  c1p[2] = { (float*)carve(CB),  (float*)carve(CB)  };
  __bf16* apad   = (__bf16*)carve((size_t)Bsz * 32 * sizeof(__bf16));
  float*  vst    = (float*)carve((size_t)Bsz * Fdim * sizeof(float));
  float*  xst    = (float*)carve((size_t)Bsz * Fdim * sizeof(float));

  float* out  = (float*)d_out;
  float* out2 = out + (size_t)PREDN * Bsz * Fdim;

  // --- init: zero state buffers + apad pad-region, convert weights, combine biases ---
  k_zero_u32<<<(Bsz * Hdim / 2 + 255) / 256, 256, 0, stream>>>((uint32_t*)hz, Bsz * Hdim / 2);
  k_zero_u32<<<(Bsz * Hdim + 255) / 256, 256, 0, stream>>>((uint32_t*)cz, Bsz * Hdim);
  k_zero_u32<<<(Bsz * 32 / 2 + 255) / 256, 256, 0, stream>>>((uint32_t*)apad, Bsz * 32 / 2);

  const int NW = FourH * Hdim;
  k_cvt<<<NW / 256, 256, 0, stream>>>(vWhh0b, vWhh0, NW);
  k_cvt<<<NW / 256, 256, 0, stream>>>(vWih1b, vWih1, NW);
  k_cvt<<<NW / 256, 256, 0, stream>>>(vWhh1b, vWhh1, NW);
  k_cvt<<<NW / 256, 256, 0, stream>>>(aWhh0b, aWhh0, NW);
  k_cvt<<<NW / 256, 256, 0, stream>>>(aWih1b, aWih1, NW);
  k_cvt<<<NW / 256, 256, 0, stream>>>(aWhh1b, aWhh1, NW);
  k_pad_wih<<<(FourH * 32) / 256, 256, 0, stream>>>(vWih0p, vWih0);
  k_pad_wih<<<(FourH * 32) / 256, 256, 0, stream>>>(aWih0p, aWih0);
  k_bias2<<<FourH / 256, 256, 0, stream>>>(vb0, vbih0, vbhh0);
  k_bias2<<<FourH / 256, 256, 0, stream>>>(vb1, vbih1, vbhh1);
  k_bias2<<<FourH / 256, 256, 0, stream>>>(ab0, abih0, abhh0);
  k_bias2<<<FourH / 256, 256, 0, stream>>>(ab1, abih1, abhh1);
  k_build_seq<<<(OBSN * Bsz * 32) / 256, 256, 0, stream>>>(vseq, v_in, 0);
  k_build_seq<<<((OBSN - 1) * Bsz * 32) / 256, 256, 0, stream>>>(aseq, v_in, 1);
  k_transpose_y<<<(PREDN * Bsz * Fdim) / 256, 256, 0, stream>>>(y, sx, out2);

  const dim3 grid(Bsz / 128, Hdim / 32);
  const int HG = (Bsz * Fdim) / 256;

  // --- v-branch LSTM (20 steps x 2 layers) ---
  {
    const __bf16* h0r = hz; const float* c0r = cz; int p0 = 0;
    const __bf16* h1r = hz; const float* c1r = cz; int p1 = 0;
    for (int t = 0; t < OBSN; ++t) {
      __bf16* h0w = h0p[p0]; float* c0w = c0p[p0];
      lstm_step<<<grid, 256, 0, stream>>>(vseq + (size_t)t * Bsz * 32, 32, vWih0p,
                                          h0r, c0r, vWhh0b, vb0, h0w, c0w);
      __bf16* h1w = h1p[p1]; float* c1w = c1p[p1];
      lstm_step<<<grid, 256, 0, stream>>>(h0w, Hdim, vWih1b,
                                          h1r, c1r, vWhh1b, vb1, h1w, c1w);
      h0r = h0w; c0r = c0w; p0 ^= 1;
      h1r = h1w; c1r = c1w; p1 ^= 1;
    }
    k_head_v<<<HG, 256, 0, stream>>>(h1r, vWl, vbl, sv, vst);
  }

  // --- a-branch LSTM (19 steps x 2 layers) ---
  {
    const __bf16* h0r = hz; const float* c0r = cz; int p0 = 0;
    const __bf16* h1r = hz; const float* c1r = cz; int p1 = 0;
    for (int t = 0; t < OBSN - 1; ++t) {
      __bf16* h0w = h0p[p0]; float* c0w = c0p[p0];
      lstm_step<<<grid, 256, 0, stream>>>(aseq + (size_t)t * Bsz * 32, 32, aWih0p,
                                          h0r, c0r, aWhh0b, ab0, h0w, c0w);
      __bf16* h1w = h1p[p1]; float* c1w = c1p[p1];
      lstm_step<<<grid, 256, 0, stream>>>(h0w, Hdim, aWih1b,
                                          h1r, c1r, aWhh1b, ab1, h1w, c1w);
      h0r = h0w; c0r = c0w; p0 ^= 1;
      h1r = h1w; c1r = c1w; p1 ^= 1;
    }
    k_head_a_init<<<HG, 256, 0, stream>>>(h1r, aWl, abl, sv, sx, x_in, vst, xst, out, apad);
  }

  // --- rollout: 29 steps, LSTM state reset to zero each step (faithful to reference) ---
  for (int s = 1; s < PREDN; ++s) {
    lstm_step<<<grid, 256, 0, stream>>>(apad, 32, aWih0p, hz, cz, aWhh0b, ab0, h0p[0], c0p[0]);
    lstm_step<<<grid, 256, 0, stream>>>(h0p[0], Hdim, aWih1b, hz, cz, aWhh1b, ab1, h1p[0], c1p[0]);
    k_head_roll<<<HG, 256, 0, stream>>>(h1p[0], aWl, abl, sv, vst, xst,
                                        out + (size_t)s * Bsz * Fdim, apad);
  }
}
